// RichNeuronUnit_49065706389520
// MI455X (gfx1250) — matrix operand order repaired and checked by reading, not compiled, not run on hardware
//
#include <hip/hip_runtime.h>

#define B_SZ 8192
#define D_SZ 256
#define N_SZ 4096

typedef __attribute__((ext_vector_type(16))) __bf16 v16bf;
typedef __attribute__((ext_vector_type(8)))  __bf16 v8bf;
typedef __attribute__((ext_vector_type(8)))  float  v8f;

#if __has_builtin(__builtin_amdgcn_global_load_async_to_lds_b128)
#define HAVE_ASYNC_LDS 1
#endif

#ifdef HAVE_ASYNC_LDS
#define AS1 __attribute__((address_space(1)))
#define AS3 __attribute__((address_space(3)))
typedef int v4i_vs __attribute__((vector_size(16)));   // matches builtin param type
__device__ __forceinline__ void async_copy16(const void* g, void* l) {
    __builtin_amdgcn_global_load_async_to_lds_b128(
        (AS1 v4i_vs*)g, (AS3 v4i_vs*)l, 0, 0);
}
__device__ __forceinline__ void async_wait0() {
#if __has_builtin(__builtin_amdgcn_s_wait_asynccnt)
    __builtin_amdgcn_s_wait_asynccnt(0);
#else
    asm volatile("s_wait_asynccnt 0x0" ::: "memory");
#endif
}
#endif

// ---- d_out layout (float elements, concatenated in return order) ----
#define O_OUT 0ull
#define O_MEM ((size_t)B_SZ * N_SZ)
#define O_TH  (O_MEM + N_SZ)
#define O_FAT (O_TH + N_SZ)
#define O_TR  (O_FAT + N_SZ)
#define O_FR  (O_TR + (size_t)N_SZ * D_SZ)

// ---- workspace layout (bytes); total ~8.5 MB ----
#define WS_XB   0ull
#define WS_WT   (WS_XB + (size_t)B_SZ * D_SZ * 2)      // bf16 x           [B][D]
#define WS_MIP  (WS_WT + (size_t)N_SZ * D_SZ * 2)      // bf16 W^T K-major [D][N]
#define WS_MOP  (WS_MIP + 64ull * D_SZ * 4)            // mean_input partials [64][D]
#define WS_MMP  (WS_MOP + 64ull * N_SZ * 4)            // mean_output partials [64][N]
#define WS_MOM  (WS_MMP + 64ull * N_SZ * 4)            // membrane partials   [64][N]
#define WS_MIM  (WS_MOM + (size_t)N_SZ * 4)            // mo_mean [N]; mi_mean [D] after

// ---------------- kernel 1: convert x to bf16 + column partial sums ----------
__global__ void __launch_bounds__(256)
k_prep_x(const float* __restrict__ x, __bf16* __restrict__ xb,
         float* __restrict__ mi_part) {
    const int d    = threadIdx.x;      // 0..255 (column)
    const int blk  = blockIdx.x;       // 0..63
    const int row0 = blk * 128;
    float s = 0.0f;
    for (int r = 0; r < 128; ++r) {
        float v = x[(size_t)(row0 + r) * D_SZ + d];
        s += v;
        xb[(size_t)(row0 + r) * D_SZ + d] = (__bf16)v;
    }
    mi_part[blk * D_SZ + d] = s;       // deterministic per-block partial
}

// ---------------- kernel 2: convert W to bf16, store K-major (transposed) ----
__global__ void __launch_bounds__(256)
k_prep_w(const float* __restrict__ W, __bf16* __restrict__ wt_g) {
    int idx = blockIdx.x * 256 + threadIdx.x;   // grid covers N*D exactly
    int n = idx >> 8;                           // row of W (D_SZ == 256)
    int d = idx & (D_SZ - 1);
    wt_g[(size_t)d * N_SZ + n] = (__bf16)W[idx];
}

// ---------------- kernel 3: WMMA GEMM + fused spiking + column partials ------
// grid = (N/128, B/128), block = 256 (8 waves), dynamic LDS = 128 KB
__global__ void __launch_bounds__(256)
k_main(const __bf16* __restrict__ xb, const __bf16* __restrict__ wt_g,
       const float* __restrict__ membrane, const float* __restrict__ threshold,
       const float* __restrict__ fatigue, float* __restrict__ out,
       float* __restrict__ mo_part, float* __restrict__ mm_part) {
    extern __shared__ __align__(16) char smem[];
    __bf16* xs = (__bf16*)smem;                  // [128][256] row-major (M x K)
    __bf16* wt = (__bf16*)(smem + 65536);        // [256][128] K-major  (K x Nloc)

    const int tid  = threadIdx.x;
    const int lane = tid & 31;
    const int wave = tid >> 5;
    const int wm   = wave >> 1;                  // 0..3 : 32-row slice
    const int wn   = wave & 1;                   // 0..1 : 64-col slice
    const int m0   = blockIdx.y * 128;
    const int n0   = blockIdx.x * 128;

    // ---- stage both tiles: byte-identical 16B copies (async direct-to-LDS) --
#ifdef HAVE_ASYNC_LDS
    #pragma unroll
    for (int it = 0; it < 16; ++it) {
        int i = tid + 256 * it;                  // x tile: 128 rows x 32 x 16B
        async_copy16(xb + (size_t)m0 * D_SZ + i * 8, xs + i * 8);
    }
    #pragma unroll
    for (int it = 0; it < 16; ++it) {
        int i = tid + 256 * it;                  // W^T tile: 256 rows x 16 x 16B
        int k = i >> 4;
        int c = i & 15;
        async_copy16(wt_g + (size_t)k * N_SZ + n0 + c * 8, wt + k * 128 + c * 8);
    }
    async_wait0();
#else
    {
        const uint4* src = (const uint4*)(xb + (size_t)m0 * D_SZ);
        uint4* dst = (uint4*)xs;
        #pragma unroll
        for (int it = 0; it < 16; ++it) dst[tid + 256 * it] = src[tid + 256 * it];
    }
    #pragma unroll
    for (int it = 0; it < 16; ++it) {
        int i = tid + 256 * it;
        int k = i >> 4;
        int c = i & 15;
        *(uint4*)(wt + k * 128 + c * 8) =
            *(const uint4*)(wt_g + (size_t)k * N_SZ + n0 + c * 8);
    }
#endif
    __syncthreads();

    v8f acc[2][4];
    #pragma unroll
    for (int mi = 0; mi < 2; ++mi)
        #pragma unroll
        for (int ni = 0; ni < 4; ++ni)
            #pragma unroll
            for (int r = 0; r < 8; ++r) acc[mi][ni][r] = 0.0f;

    const int mrow0  = wm * 32 + (lane & 15);
    const int ka_off = (lane >> 4) << 3;         // A lane-half K offset (0 / 8)

    for (int k0 = 0; k0 < D_SZ; k0 += 32) {
        v16bf a[2];
        #pragma unroll
        for (int mi = 0; mi < 2; ++mi) {
            const __bf16* p = xs + (mrow0 + 16 * mi) * 256 + k0 + ka_off;
            v8bf lo = *(const v8bf*)p;           // K = base..base+7
            v8bf hi = *(const v8bf*)(p + 16);    // K = base+16..base+23
            a[mi] = __builtin_shufflevector(lo, hi,
                     0,1,2,3,4,5,6,7,8,9,10,11,12,13,14,15);
        }
        #pragma unroll
        for (int ni = 0; ni < 4; ++ni) {
            // B lane = N column (lane&15), elements = K in table order (same as A)
            v16bf b;
            #pragma unroll
            for (int s = 0; s < 16; ++s)
                b[s] = wt[(k0 + ka_off + (s & 7) + ((s >> 3) << 4)) * 128 + wn * 64 + ni * 16 + (lane & 15)];
            #pragma unroll
            for (int mi = 0; mi < 2; ++mi) {
                acc[mi][ni] = __builtin_amdgcn_wmma_f32_16x16x32_bf16(
                    false, a[mi], false, b, (short)0, acc[mi][ni], false, false);
            }
        }
    }
    __syncthreads();                             // tiles consumed; reuse LDS

    float* redO = (float*)smem;                  // [8 waves][128 cols]
    float* redM = redO + 8 * 128;

    const int laneHi = lane >> 4;
    #pragma unroll
    for (int ni = 0; ni < 4; ++ni) {
        const int nloc = wn * 64 + ni * 16 + (lane & 15);
        const int n    = n0 + nloc;
        const float th  = threshold[n];
        const float fat = fatigue[n];
        const float mbr = membrane[n];
        float so = 0.0f, sm = 0.0f;
        #pragma unroll
        for (int mi = 0; mi < 2; ++mi) {
            #pragma unroll
            for (int r = 0; r < 8; ++r) {
                float syn = acc[mi][ni][r];
                float mem = 0.9f * mbr + 0.1f * (syn - th);
                float prob = (1.0f - fat) / (1.0f + __expf(-mem));
                float sgn = (mem > 0.0f) ? 1.0f : ((mem < 0.0f) ? -1.0f : 0.0f);
                float o = (prob > 0.5f) ? sgn : 0.0f;
                int m = m0 + wm * 32 + mi * 16 + laneHi * 8 + r;
                __builtin_nontemporal_store(o, out + O_OUT + (size_t)m * N_SZ + n);
                so += o;
                sm += mem;
            }
        }
        // fold the other half-lane (same column, rows +8)
        so += __shfl_down(so, 16, 32);
        sm += __shfl_down(sm, 16, 32);
        if (laneHi == 0) {
            redO[wave * 128 + nloc] = so;
            redM[wave * 128 + nloc] = sm;
        }
    }
    __syncthreads();

    if (tid < 128) {                             // fixed-order wave reduction
        const int c = tid;
        const int half = c >> 6;                 // which wn owns this column
        float so = 0.0f, sm = 0.0f;
        #pragma unroll
        for (int w = 0; w < 4; ++w) {
            so += redO[(2 * w + half) * 128 + c];
            sm += redM[(2 * w + half) * 128 + c];
        }
        mo_part[(size_t)blockIdx.y * N_SZ + n0 + c] = so;
        mm_part[(size_t)blockIdx.y * N_SZ + n0 + c] = sm;
    }
}

// ---------------- kernel 4: finalize per-neuron state ------------------------
__global__ void __launch_bounds__(256)
k_finalize(const float* __restrict__ mo_part, const float* __restrict__ mm_part,
           const float* __restrict__ mi_part, const float* __restrict__ threshold,
           const float* __restrict__ fatigue, const float* __restrict__ firing_rate,
           float* __restrict__ out, float* __restrict__ mo_mean,
           float* __restrict__ mi_mean) {
    int n = blockIdx.x * 256 + threadIdx.x;      // 0..N-1 (grid = 16)
    float so = 0.0f, sm = 0.0f;
    for (int j = 0; j < 64; ++j) {               // fixed order: deterministic
        so += mo_part[(size_t)j * N_SZ + n];
        sm += mm_part[(size_t)j * N_SZ + n];
    }
    const float inv_b = 1.0f / (float)B_SZ;
    float mo  = so * inv_b;
    float amo = fabsf(mo);
    out[O_MEM + n] = sm * inv_b;
    out[O_TH  + n] = 0.99f * threshold[n] + 0.01f * (1.0f + 0.1f * amo);
    float nf = fatigue[n] - 0.1f + 0.5f * amo;
    out[O_FAT + n] = fminf(fmaxf(nf, 0.0f), 0.9f);
    out[O_FR  + n] = 0.99f * firing_rate[n] + 0.01f * fminf(amo, 1.0f);
    mo_mean[n] = mo;
    if (n < D_SZ) {
        float si = 0.0f;
        for (int j = 0; j < 64; ++j) si += mi_part[j * D_SZ + n];
        mi_mean[n] = si * inv_b;
    }
}

// ---------------- kernel 5: trace rank-1 update ------------------------------
__global__ void __launch_bounds__(256)
k_trace(const float* __restrict__ trace, const float* __restrict__ mo_mean,
        const float* __restrict__ mi_mean, float* __restrict__ out_tr) {
    int idx = blockIdx.x * 256 + threadIdx.x;    // grid = N*D/256 = 4096
    int n = idx >> 8;
    int d = idx & (D_SZ - 1);
    float t = __builtin_nontemporal_load(trace + idx);
    out_tr[idx] = 0.95f * t + mo_mean[n] * mi_mean[d];
}

// ---------------- host launcher ----------------------------------------------
extern "C" void kernel_launch(void* const* d_in, const int* in_sizes, int n_in,
                              void* d_out, int out_size, void* d_ws, size_t ws_size,
                              hipStream_t stream) {
    const float* x           = (const float*)d_in[0];
    const float* W           = (const float*)d_in[1];
    const float* membrane    = (const float*)d_in[2];
    const float* threshold   = (const float*)d_in[3];
    const float* fatigue     = (const float*)d_in[4];
    const float* trace       = (const float*)d_in[5];
    const float* firing_rate = (const float*)d_in[6];
    float* out = (float*)d_out;
    char*  ws  = (char*)d_ws;

    __bf16* xb      = (__bf16*)(ws + WS_XB);
    __bf16* wt_g    = (__bf16*)(ws + WS_WT);
    float*  mi_part = (float*)(ws + WS_MIP);
    float*  mo_part = (float*)(ws + WS_MOP);
    float*  mm_part = (float*)(ws + WS_MMP);
    float*  mo_mean = (float*)(ws + WS_MOM);
    float*  mi_mean = (float*)(ws + WS_MIM + N_SZ * 4);

    k_prep_x<<<64, 256, 0, stream>>>(x, xb, mi_part);
    k_prep_w<<<(N_SZ * D_SZ) / 256, 256, 0, stream>>>(W, wt_g);

    dim3 g3(N_SZ / 128, B_SZ / 128);             // (32, 64)
    k_main<<<g3, 256, 128 * 1024, stream>>>(xb, wt_g, membrane, threshold, fatigue,
                                            out, mo_part, mm_part);

    k_finalize<<<N_SZ / 256, 256, 0, stream>>>(mo_part, mm_part, mi_part,
                                               threshold, fatigue, firing_rate,
                                               out, mo_mean, mi_mean);

    k_trace<<<(N_SZ * D_SZ) / 256, 256, 0, stream>>>(trace, mo_mean, mi_mean,
                                                     out + O_TR);
    (void)in_sizes; (void)n_in; (void)out_size; (void)ws_size; (void)membrane;
}
